// MambaLayer_3633542332579
// MI455X (gfx1250) — compile-verified, matrix operand order checked by reading, not run on hardware
//
#include <hip/hip_runtime.h>
#include <stdint.h>

// ---------------- Problem constants (match reference) ----------------
#define BATCH   2
#define SEQL    1024
#define DMODEL  1024
#define DINNER  2048
#define DCONV   4
#define DTRANK  64
#define NSTATE  16
#define MROWS   (BATCH * SEQL)        // 2048 token rows
#define XZCOLS  (2 * DINNER)          // 4096

// ---------------- WMMA types ----------------
typedef __attribute__((ext_vector_type(16))) __bf16 v16bf;
typedef __attribute__((ext_vector_type(8)))  float  v8f;

union FragBF16 {
    uint4 u4[2];
    v16bf v;
};

__device__ __forceinline__ uint16_t f32_to_bf16(float f) {
    uint32_t u = __float_as_uint(f);
    uint32_t r = u + 0x7FFFu + ((u >> 16) & 1u);   // round-to-nearest-even
    return (uint16_t)(r >> 16);
}

// ---------------- f32 -> bf16 weight conversion ----------------
__global__ void cvt_f32_bf16_kernel(const float* __restrict__ in,
                                    uint16_t* __restrict__ out, int n) {
    int i = blockIdx.x * blockDim.x + threadIdx.x;
    if (i < n) out[i] = f32_to_bf16(in[i]);
}

// ---------------- LayerNorm over DMODEL, emit bf16 ----------------
__global__ void ln_kernel(const float* __restrict__ x,
                          const float* __restrict__ w,
                          const float* __restrict__ b,
                          uint16_t* __restrict__ xn) {
    const int row = blockIdx.x;                 // 0..MROWS-1
    const int tid = threadIdx.x;                // 256 threads
    const float* xr = x + (size_t)row * DMODEL;

    float v[4];
    float s = 0.f, s2 = 0.f;
#pragma unroll
    for (int i = 0; i < 4; ++i) {
        float t = xr[tid + 256 * i];
        v[i] = t; s += t; s2 += t * t;
    }
    // wave32 reduction
#pragma unroll
    for (int o = 16; o > 0; o >>= 1) {
        s  += __shfl_xor(s,  o, 32);
        s2 += __shfl_xor(s2, o, 32);
    }
    __shared__ float sh[16];
    const int wid = tid >> 5;                   // 8 waves per block
    if ((tid & 31) == 0) { sh[wid] = s; sh[8 + wid] = s2; }
    __syncthreads();
    if (tid < 32) {
        float a  = (tid < 8) ? sh[tid]     : 0.f;
        float a2 = (tid < 8) ? sh[8 + tid] : 0.f;
#pragma unroll
        for (int o = 4; o > 0; o >>= 1) {
            a  += __shfl_xor(a,  o, 32);
            a2 += __shfl_xor(a2, o, 32);
        }
        if (tid == 0) { sh[0] = a; sh[1] = a2; }
    }
    __syncthreads();
    const float mu   = sh[0] * (1.f / DMODEL);
    const float var  = sh[1] * (1.f / DMODEL) - mu * mu;
    const float rstd = rsqrtf(var + 1e-5f);
#pragma unroll
    for (int i = 0; i < 4; ++i) {
        int c = tid + 256 * i;
        float o = (v[i] - mu) * rstd * w[c] + b[c];
        xn[(size_t)row * DMODEL + c] = f32_to_bf16(o);
    }
}

// ---------------- Generic bf16 WMMA GEMM: C[M,N] = A[M,K] * B[N,K]^T --------
// A: row-major [M,K] bf16; B: row-major [N,K] bf16 (weight layout); C: f32 [M,N].
// Block = 4 waves; each wave owns a 16(M) x (NT*16)(N) tile.
// K processed in chunks of 64 with explicit double buffering: loads for chunk
// k+32 / k+64 are in flight while WMMAs consume chunk k -> partial loadcnt
// waits instead of full drains. Requires K % 64 == 0 (all call sites comply).
// Epilogue over-reads <=190B past the row region; every operand buffer is
// followed by another workspace carve, so this stays in-bounds of d_ws.
// Per-lane fragment layout per CDNA5 ISA 7.12.2 (16-bit A 16x32):
//   lanes 0-15  (row = lane)     hold K {0..7, 16..23}
//   lanes 16-31 (row = lane-16)  hold K {8..15, 24..31}
template <int NT>
__global__ void wmma_gemm_bf16_kernel(const uint16_t* __restrict__ A,
                                      const uint16_t* __restrict__ B,
                                      float* __restrict__ C,
                                      int N, int K, int n_off) {
    const int lane = threadIdx.x & 31;
    const int wave = threadIdx.x >> 5;          // 4 waves per block
    const int half = lane >> 4;                 // 0 or 1
    const int lr   = lane & 15;
    const int m0   = (blockIdx.x * 4 + wave) * 16;
    const int n0   = blockIdx.y * (NT * 16) + n_off;
    const int klo  = half ? 8 : 0;

    const uint16_t* ap = A + (size_t)(m0 + lr) * K + klo;
    const uint16_t* bp[NT];
#pragma unroll
    for (int t = 0; t < NT; ++t)
        bp[t] = B + (size_t)(n0 + t * 16 + lr) * K + klo;

    v8f acc[NT] = {};
    FragBF16 a0, a1, b0[NT], b1[NT];

    // prologue: chunk 0 in flight
    a0.u4[0] = *(const uint4*)(ap);
    a0.u4[1] = *(const uint4*)(ap + 16);
#pragma unroll
    for (int t = 0; t < NT; ++t) {
        b0[t].u4[0] = *(const uint4*)(bp[t]);
        b0[t].u4[1] = *(const uint4*)(bp[t] + 16);
    }

    for (int k = 0; k < K; k += 64) {
        // issue loads for chunk k+32 before consuming chunk k
        a1.u4[0] = *(const uint4*)(ap + k + 32);
        a1.u4[1] = *(const uint4*)(ap + k + 48);
#pragma unroll
        for (int t = 0; t < NT; ++t) {
            b1[t].u4[0] = *(const uint4*)(bp[t] + k + 32);
            b1[t].u4[1] = *(const uint4*)(bp[t] + k + 48);
        }
        __builtin_prefetch(ap + k + 256, 0, 1);  // speculative, unconditional
#pragma unroll
        for (int t = 0; t < NT; ++t)
            acc[t] = __builtin_amdgcn_wmma_f32_16x16x32_bf16(
                false, a0.v, false, b0[t].v, (short)0, acc[t], false, false);

        // issue loads for chunk k+64 before consuming chunk k+32
        a0.u4[0] = *(const uint4*)(ap + k + 64);
        a0.u4[1] = *(const uint4*)(ap + k + 80);
#pragma unroll
        for (int t = 0; t < NT; ++t) {
            b0[t].u4[0] = *(const uint4*)(bp[t] + k + 64);
            b0[t].u4[1] = *(const uint4*)(bp[t] + k + 80);
        }
#pragma unroll
        for (int t = 0; t < NT; ++t)
            acc[t] = __builtin_amdgcn_wmma_f32_16x16x32_bf16(
                false, a1.v, false, b1[t].v, (short)0, acc[t], false, false);
    }

    // C/D layout: lanes 0-15: vgpr r -> M=r, N=lane; lanes 16-31: M=r+8, N=lane-16
#pragma unroll
    for (int t = 0; t < NT; ++t) {
        const int col = n0 + t * 16 + lr;
#pragma unroll
        for (int r = 0; r < 8; ++r) {
            const int row = m0 + r + half * 8;
            C[(size_t)row * N + col] = acc[t][r];
        }
    }
}

// ---------------- Depthwise causal conv1d (k=4) + SiLU ----------------
// Input: xs part of xz buffer [MROWS, XZCOLS], cols 0..DINNER-1.
// Output: xs f32 (for scan) and bf16 (for W_x WMMA GEMM).
__global__ void conv_silu_kernel(const float* __restrict__ xz,
                                 const float* __restrict__ cw,
                                 const float* __restrict__ cb,
                                 float* __restrict__ xs_f32,
                                 uint16_t* __restrict__ xs_bf) {
    int idx = blockIdx.x * blockDim.x + threadIdx.x;     // m * DINNER + c
    if (idx >= MROWS * DINNER) return;
    const int c = idx & (DINNER - 1);
    const int m = idx >> 11;                             // / DINNER
    const int l = m & (SEQL - 1);
    float acc = cb[c];
#pragma unroll
    for (int k = 0; k < DCONV; ++k) {
        int dl = l - (DCONV - 1) + k;
        if (dl >= 0)
            acc += cw[c * DCONV + k] *
                   xz[(size_t)(m - (DCONV - 1) + k) * XZCOLS + c];
    }
    float s = acc / (1.f + __expf(-acc));                // SiLU
    xs_f32[idx] = s;
    xs_bf[idx]  = f32_to_bf16(s);
}

// ---------------- Extract dt-rank slice of xdbl as bf16 ----------------
__global__ void dt_prep_kernel(const float* __restrict__ xdbl,
                               uint16_t* __restrict__ dtr_bf) {
    int i = blockIdx.x * blockDim.x + threadIdx.x;       // m * DTRANK + r
    if (i >= MROWS * DTRANK) return;
    int m = i >> 6, r = i & 63;
    dtr_bf[i] = f32_to_bf16(xdbl[(size_t)m * 96 + r]);
}

// ---------------- softplus(dt_raw + b_dt), in place ----------------
__global__ void softplus_kernel(float* __restrict__ dt,
                                const float* __restrict__ b_dt) {
    int i = blockIdx.x * blockDim.x + threadIdx.x;
    if (i >= MROWS * DINNER) return;
    float v = dt[i] + b_dt[i & (DINNER - 1)];
    dt[i] = (v > 20.f) ? v : __logf(1.f + __expf(v));
}

// ---------------- Fused selective scan + SiLU gate ----------------
// One thread per (batch, channel). 16-state recurrence kept in VGPRs.
// B/C rows (32 f32 per step) staged in LDS per 64-step tile (broadcast reads).
__global__ void scan_kernel(const float* __restrict__ dt,
                            const float* __restrict__ xs,
                            const float* __restrict__ xdbl,
                            const float* __restrict__ xz,
                            const float* __restrict__ A_log,
                            const float* __restrict__ Dp,
                            uint16_t* __restrict__ y_bf) {
    __shared__ float bc[64 * 32];                         // 8 KB tile of B|C
    const int b = blockIdx.x >> 3;                        // 8 blocks per batch
    const int c = (blockIdx.x & 7) * 256 + threadIdx.x;   // blockDim = 256
    float An[NSTATE], h[NSTATE];
#pragma unroll
    for (int n = 0; n < NSTATE; ++n) {
        An[n] = -__expf(A_log[c * NSTATE + n]);          // A = -exp(A_log)
        h[n]  = 0.f;
    }
    const float Dc = Dp[c];

    for (int l0 = 0; l0 < SEQL; l0 += 64) {
        __syncthreads();
        for (int i = threadIdx.x; i < 64 * 32; i += 256) {
            int st = i >> 5, col = i & 31;               // B at 64..79, C at 80..95
            bc[i] = xdbl[(size_t)(b * SEQL + l0 + st) * 96 + 64 + col];
        }
        __syncthreads();
        for (int s = 0; s < 64; ++s) {
            const size_t idx = (size_t)(b * SEQL + l0 + s) * DINNER + c;
            const float dtv = dt[idx];
            const float xv  = xs[idx];
            const float dbx = dtv * xv;
            float y = Dc * xv;
#pragma unroll
            for (int n = 0; n < NSTATE; ++n) {
                float dA = __expf(dtv * An[n]);          // v_exp_f32
                h[n] = dA * h[n] + dbx * bc[s * 32 + n];
                y += h[n] * bc[s * 32 + 16 + n];
            }
            const float zv = xz[(size_t)(b * SEQL + l0 + s) * XZCOLS + DINNER + c];
            const float g  = zv / (1.f + __expf(-zv));   // SiLU gate
            y_bf[idx] = f32_to_bf16(y * g);
        }
    }
}

// ---------------- residual add ----------------
__global__ void residual_kernel(const float* __restrict__ x,
                                const float* __restrict__ o,
                                float* __restrict__ out, int n) {
    int i = blockIdx.x * blockDim.x + threadIdx.x;
    if (i < n) out[i] = x[i] + o[i];
}

// ---------------- host launcher ----------------
extern "C" void kernel_launch(void* const* d_in, const int* in_sizes, int n_in,
                              void* d_out, int out_size, void* d_ws, size_t ws_size,
                              hipStream_t stream) {
    const float* x      = (const float*)d_in[0];
    const float* ln_w   = (const float*)d_in[1];
    const float* ln_b   = (const float*)d_in[2];
    const float* W_in   = (const float*)d_in[3];
    const float* conv_w = (const float*)d_in[4];
    const float* conv_b = (const float*)d_in[5];
    const float* W_x    = (const float*)d_in[6];
    const float* W_dt   = (const float*)d_in[7];
    const float* b_dt   = (const float*)d_in[8];
    const float* A_log  = (const float*)d_in[9];
    const float* Dp     = (const float*)d_in[10];
    const float* W_out  = (const float*)d_in[11];
    float*       out    = (float*)d_out;

    // workspace carve-out (256B aligned regions)
    char* p = (char*)d_ws;
    auto carve = [&](size_t bytes) -> char* {
        char* r = p;
        p += (bytes + 255) & ~(size_t)255;
        return r;
    };
    uint16_t* xn_bf   = (uint16_t*)carve((size_t)MROWS * DMODEL  * 2);
    uint16_t* Win_bf  = (uint16_t*)carve((size_t)XZCOLS * DMODEL * 2);
    uint16_t* Wx_bf   = (uint16_t*)carve((size_t)96 * DINNER     * 2);
    uint16_t* Wdt_bf  = (uint16_t*)carve((size_t)DINNER * DTRANK * 2);
    uint16_t* Wout_bf = (uint16_t*)carve((size_t)DMODEL * DINNER * 2);
    float*    xz      = (float*)   carve((size_t)MROWS * XZCOLS  * 4);
    float*    xs_f32  = (float*)   carve((size_t)MROWS * DINNER  * 4);
    uint16_t* xs_bf   = (uint16_t*)carve((size_t)MROWS * DINNER  * 2);
    float*    xdbl    = (float*)   carve((size_t)MROWS * 96      * 4);
    uint16_t* dtr_bf  = (uint16_t*)carve((size_t)MROWS * DTRANK  * 2);
    float*    dtbuf   = (float*)   carve((size_t)MROWS * DINNER  * 4);
    uint16_t* y_bf    = (uint16_t*)carve((size_t)MROWS * DINNER  * 2);
    float*    outraw  = (float*)   carve((size_t)MROWS * DMODEL  * 4);

    const int T = 256;
    auto grid1 = [](int n, int t) { return (n + t - 1) / t; };

    // weight conversion f32 -> bf16 (cheap, done every call: deterministic)
    cvt_f32_bf16_kernel<<<grid1(XZCOLS * DMODEL, T), T, 0, stream>>>(W_in,  Win_bf,  XZCOLS * DMODEL);
    cvt_f32_bf16_kernel<<<grid1(96 * DINNER,     T), T, 0, stream>>>(W_x,   Wx_bf,   96 * DINNER);
    cvt_f32_bf16_kernel<<<grid1(DINNER * DTRANK, T), T, 0, stream>>>(W_dt,  Wdt_bf,  DINNER * DTRANK);
    cvt_f32_bf16_kernel<<<grid1(DMODEL * DINNER, T), T, 0, stream>>>(W_out, Wout_bf, DMODEL * DINNER);

    // 1. LayerNorm -> bf16
    ln_kernel<<<MROWS, 256, 0, stream>>>(x, ln_w, ln_b, xn_bf);

    // 2. in_proj GEMM: xz[2048,4096] = xn * W_in^T   (K=1024)
    wmma_gemm_bf16_kernel<4><<<dim3(MROWS / 64, XZCOLS / 64), 128, 0, stream>>>(
        xn_bf, Win_bf, xz, XZCOLS, DMODEL, 0);

    // 3. depthwise causal conv + SiLU
    conv_silu_kernel<<<grid1(MROWS * DINNER, T), T, 0, stream>>>(
        xz, conv_w, conv_b, xs_f32, xs_bf);

    // 4. x_dbl GEMM: xdbl[2048,96] = xs * W_x^T  (K=2048; 96 = 64 + 32 cols)
    wmma_gemm_bf16_kernel<4><<<dim3(MROWS / 64, 1), 128, 0, stream>>>(
        xs_bf, Wx_bf, xdbl, 96, DINNER, 0);
    wmma_gemm_bf16_kernel<2><<<dim3(MROWS / 64, 1), 128, 0, stream>>>(
        xs_bf, Wx_bf, xdbl, 96, DINNER, 64);

    // 5. dt GEMM: dt_raw[2048,2048] = dt_rank * W_dt^T  (K=64)
    dt_prep_kernel<<<grid1(MROWS * DTRANK, T), T, 0, stream>>>(xdbl, dtr_bf);
    wmma_gemm_bf16_kernel<4><<<dim3(MROWS / 64, DINNER / 64), 128, 0, stream>>>(
        dtr_bf, Wdt_bf, dtbuf, DINNER, DTRANK, 0);
    softplus_kernel<<<grid1(MROWS * DINNER, T), T, 0, stream>>>(dtbuf, b_dt);

    // 6. fused selective scan + gate -> y bf16
    scan_kernel<<<BATCH * (DINNER / 256), 256, 0, stream>>>(
        dtbuf, xs_f32, xdbl, xz, A_log, Dp, y_bf);

    // 7. out_proj GEMM: outraw[2048,1024] = y * W_out^T  (K=2048)
    wmma_gemm_bf16_kernel<4><<<dim3(MROWS / 64, DMODEL / 64), 128, 0, stream>>>(
        y_bf, Wout_bf, outraw, DMODEL, DINNER, 0);

    // 8. residual add
    residual_kernel<<<grid1(MROWS * DMODEL, T), T, 0, stream>>>(
        x, outraw, out, MROWS * DMODEL);
}
